// ParameterizedAttention_163208757637
// MI455X (gfx1250) — compile-verified, matrix operand order checked by reading, not run on hardware
//
#include <hip/hip_runtime.h>

// ParameterizedAttention for MI455X (gfx1250, wave32, WMMA bf16).
// B=32 T=256 J=25(pad 32) C=256 H=64 ; one workgroup per (b,t); 8 waves.

#define BB 32
#define TT 256
#define JJ 25
#define CC 256
#define HH 64
#define JP 32
#define NBLK (BB * TT)

typedef __attribute__((ext_vector_type(16))) __bf16 v16bf;
typedef __attribute__((ext_vector_type(8)))  float  v8f;

static __device__ __forceinline__ unsigned short f2bf(float f) {
  unsigned u = __builtin_bit_cast(unsigned, f);
  u = (u + 0x7fffu + ((u >> 16) & 1u)) >> 16;   // round-to-nearest-even
  return (unsigned short)u;
}

// Build a 16x32 bf16 A/B fragment (K contiguous in LDS rows).
// Layout per ISA 7.12.2 (16-bit A 16x32): lanes 0-15 / 16-31 both hold rows
// 0-15; VGPR p, elems (2p,2p+1): K = (p>=4?16:0) + 8*(lane/16) + 2*(p&3) + e.
static __device__ __forceinline__ v16bf frag_k(const unsigned short* base,
                                               int stride, int row, int k0, int h) {
  v16bf f;
  const unsigned short* r = base + row * stride + k0 + (h << 3);
#pragma unroll
  for (int p = 0; p < 8; ++p) {
    int kb = ((p & 4) << 2) + ((p & 3) << 1);     // even -> 4B aligned
    unsigned u = *(const unsigned*)(r + kb);
    f[2 * p]     = __builtin_bit_cast(__bf16, (unsigned short)(u & 0xffffu));
    f[2 * p + 1] = __builtin_bit_cast(__bf16, (unsigned short)(u >> 16));
  }
  return f;
}

#define WMMA(a, b, c) \
  __builtin_amdgcn_wmma_f32_16x16x32_bf16(false, (a), false, (b), (short)0, (c), false, false)

__global__ __launch_bounds__(256) void pattn_kernel(
    const float* __restrict__ x, const float* __restrict__ bias_mat,
    const float* __restrict__ Wq, const float* __restrict__ bq,
    const float* __restrict__ Wk, const float* __restrict__ bk,
    const float* __restrict__ Wv, const float* __restrict__ bv,
    const float* __restrict__ attn_w, const float* __restrict__ att_bias,
    const float* __restrict__ ln_g, const float* __restrict__ ln_b,
    float* __restrict__ out) {
  // ---- LDS (bf16 stored as ushort; padded strides vs bank conflicts) ----
  __shared__ unsigned short xs[JP * CC];        // x, bf16, rows padded to 32
  __shared__ unsigned short w0[64 * 260];       // Wq^T, later Wv-chunk^T (N x K)
  __shared__ unsigned short w1[64 * 260];       // Wk^T
  __shared__ unsigned short awsT[JP * 68];      // attn_w^T (J x H), J padded
  __shared__ unsigned short qs[JP * HH];        // q  (J x H)
  __shared__ unsigned short ks[JP * HH];        // k  (J x H)
  __shared__ float          sc[JP * JP];        // scores f32
  __shared__ unsigned short cofs[JP * JP];      // softmax coefs bf16
  __shared__ unsigned short vsT[64 * 36];       // v-chunk^T (N x J)
  __shared__ float          ybuf[JP * CC];      // attended + residual, f32

  const int  tid   = threadIdx.x;
  const int  bt    = blockIdx.x;
  const long xbase = (long)bt * (JJ * CC);
  const int  wid = tid >> 5, lane = tid & 31;
  const int  hh = lane >> 4, ln16 = lane & 15;
  const int  mi = wid & 1, ni = wid >> 1;       // wave -> (M-tile, N-tile)

  // ---- Phase A: stage x, Wq^T, Wk^T, attn_w^T as bf16 ----
  for (int i = tid; i < JP * CC; i += 256) {
    int r = i >> 8, c = i & 255;
    xs[i] = f2bf((r < JJ) ? x[xbase + r * CC + c] : 0.f);
  }
  for (int i = tid; i < CC * HH; i += 256) {
    int n = i & 63, k = i >> 6;                 // coalesced global reads
    w0[n * 260 + k] = f2bf(Wq[k * HH + n]);
    w1[n * 260 + k] = f2bf(Wk[k * HH + n]);
  }
  for (int i = tid; i < JP * HH; i += 256) {
    int j = i >> 6, h = i & 63;
    awsT[j * 68 + h] = f2bf((j < JJ) ? attn_w[h * JJ + j] : 0.f);
  }
  __syncthreads();

  // ---- Phase B: q = xWq + bq, k = xWk + bk (wave owns one 16x16 tile each) ----
  {
    v8f aq = {}, ak = {};
#pragma unroll
    for (int k0 = 0; k0 < CC; k0 += 32) {
      v16bf a  = frag_k(xs, CC, mi * 16 + ln16, k0, hh);
      v16bf b0 = frag_k(w0, 260, ni * 16 + ln16, k0, hh);
      v16bf b1 = frag_k(w1, 260, ni * 16 + ln16, k0, hh);
      aq = WMMA(a, b0, aq);
      ak = WMMA(a, b1, ak);
    }
#pragma unroll
    for (int r = 0; r < 8; ++r) {               // C/D layout: M=r+8*(lane/16)
      int m = mi * 16 + r + 8 * hh;
      int n = ni * 16 + ln16;
      qs[m * HH + n] = f2bf(aq[r] + bq[n]);
      ks[m * HH + n] = f2bf(ak[r] + bk[n]);
    }
  }
  __syncthreads();

  // ---- Phase C: scores = q@attn_w + q@k^T * scale + biases (waves 0-3) ----
  if (wid < 4) {
    int m2 = wid & 1, n2 = wid >> 1;
    v8f aw = {}, ad = {};
#pragma unroll
    for (int k0 = 0; k0 < HH; k0 += 32) {
      v16bf a  = frag_k(qs, HH, m2 * 16 + ln16, k0, hh);
      v16bf bw = frag_k(awsT, 68, n2 * 16 + ln16, k0, hh);
      v16bf bd = frag_k(ks, HH, n2 * 16 + ln16, k0, hh);   // B = k^T
      aw = WMMA(a, bw, aw);
      ad = WMMA(a, bd, ad);
    }
#pragma unroll
    for (int r = 0; r < 8; ++r) {
      int m = m2 * 16 + r + 8 * hh;
      int n = n2 * 16 + ln16;
      float s = -1e30f;                          // mask padded columns
      if (n < JJ) {
        float bm = (m < JJ) ? bias_mat[m * JJ + n] : 0.f;
        s = aw[r] + att_bias[n] + ad[r] * 0.125f + bm;  // 1/sqrt(64)
      }
      sc[m * JP + n] = s;
    }
  }
  __syncthreads();

  // ---- Phase D: softmax over 25 cols (wave 0, one row per lane) ----
  if (wid == 0) {
    int m = lane;
    float mx = -1e30f;
#pragma unroll
    for (int n = 0; n < JJ; ++n) mx = fmaxf(mx, sc[m * JP + n]);
    float sum = 0.f;
#pragma unroll
    for (int n = 0; n < JJ; ++n) sum += __expf(sc[m * JP + n] - mx);
    float inv = 1.f / sum;
#pragma unroll
    for (int n = 0; n < JP; ++n) {
      float t = (n < JJ) ? __expf(sc[m * JP + n] - mx) * inv : 0.f;
      cofs[m * JP + n] = f2bf(t);
    }
  }
  __syncthreads();

  // ---- Phase E: stream Wv in 4 x 64-col chunks; fuse v -> attended ----
  for (int ch = 0; ch < 4; ++ch) {
    for (int i = tid; i < CC * 64; i += 256) {   // stage Wv chunk^T into w0
      int n = i & 63, k = i >> 6;
      w0[n * 260 + k] = f2bf(Wv[k * CC + ch * 64 + n]);
    }
    __syncthreads();
    {                                            // v chunk: 32x64, K=256
      v8f acc = {};
#pragma unroll
      for (int k0 = 0; k0 < CC; k0 += 32) {
        v16bf a = frag_k(xs, CC, mi * 16 + ln16, k0, hh);
        v16bf b = frag_k(w0, 260, ni * 16 + ln16, k0, hh);
        acc = WMMA(a, b, acc);
      }
#pragma unroll
      for (int r = 0; r < 8; ++r) {
        int m = mi * 16 + r + 8 * hh;
        int n = ni * 16 + ln16;
        vsT[n * 36 + m] = f2bf(acc[r] + bv[ch * 64 + n]);   // store transposed
      }
    }
    __syncthreads();
    {                                            // attended chunk: K=32, 1 WMMA
      v16bf a = frag_k(cofs, JP, mi * 16 + ln16, 0, hh);
      v16bf b = frag_k(vsT, 36, ni * 16 + ln16, 0, hh);
      v8f acc = {};
      acc = WMMA(a, b, acc);
#pragma unroll
      for (int r = 0; r < 8; ++r) {
        int m = mi * 16 + r + 8 * hh;
        int n = ni * 16 + ln16;
        int cg = ch * 64 + n;
        float res = (m < JJ) ? x[xbase + m * CC + cg] : 0.f; // residual (L2 hit)
        ybuf[m * CC + cg] = acc[r] + res;
      }
    }
    __syncthreads();
  }

  // ---- Phase F: LayerNorm over C=256, wave32 shuffle reduction ----
  for (int m = wid; m < JJ; m += 8) {
    const float* row = &ybuf[m * CC];
    float s = 0.f, s2 = 0.f;
#pragma unroll
    for (int i = 0; i < 8; ++i) {
      float v = row[lane * 8 + i];
      s += v; s2 += v * v;
    }
#pragma unroll
    for (int off = 16; off > 0; off >>= 1) {
      s  += __shfl_xor(s, off, 32);
      s2 += __shfl_xor(s2, off, 32);
    }
    float mu   = s * (1.f / CC);
    float rstd = rsqrtf(s2 * (1.f / CC) - mu * mu + 1e-5f);
#pragma unroll
    for (int i = 0; i < 8; ++i) {
      int c = lane * 8 + i;
      out[xbase + m * CC + c] = (row[c] - mu) * rstd * ln_g[c] + ln_b[c];
    }
  }
}

extern "C" void kernel_launch(void* const* d_in, const int* in_sizes, int n_in,
                              void* d_out, int out_size, void* d_ws, size_t ws_size,
                              hipStream_t stream) {
  (void)in_sizes; (void)n_in; (void)out_size; (void)d_ws; (void)ws_size;
  const float* x        = (const float*)d_in[0];
  const float* bias_mat = (const float*)d_in[1];
  const float* Wq       = (const float*)d_in[2];
  const float* bq       = (const float*)d_in[3];
  const float* Wk       = (const float*)d_in[4];
  const float* bk       = (const float*)d_in[5];
  const float* Wv       = (const float*)d_in[6];
  const float* bv       = (const float*)d_in[7];
  const float* attn_w   = (const float*)d_in[8];
  const float* att_bias = (const float*)d_in[9];
  const float* ln_g     = (const float*)d_in[10];
  const float* ln_b     = (const float*)d_in[11];

  pattn_kernel<<<dim3(NBLK), dim3(256), 0, stream>>>(
      x, bias_mat, Wq, bq, Wk, bk, Wv, bv, attn_w, att_bias, ln_g, ln_b,
      (float*)d_out);
}
